// EncBlock_3487513445071
// MI455X (gfx1250) — matrix operand order repaired and checked by reading, not compiled, not run on hardware
//
#include <hip/hip_runtime.h>
#include <cstdint>
#include <cstddef>

// ---------------- problem constants ----------------
#define BB   8
#define DD   384
#define FDIM 256
#define NPIX 4096   // 64*64
#define NHD  8
#define HDD  48
#define GSZ  1024   // pixels per cloud
#define MROW 32768  // BB*NPIX
#define DFF  1536   // 4*DD

typedef __attribute__((ext_vector_type(8)))  __bf16 v8bf;
typedef __attribute__((ext_vector_type(16))) __bf16 v16bf;
typedef __attribute__((ext_vector_type(8)))  float  v8f;
typedef __attribute__((ext_vector_type(4)))  unsigned int v4u;
typedef __attribute__((ext_vector_type(8)))  int v8i;
typedef __attribute__((ext_vector_type(4)))  int v4i;

static __device__ __forceinline__ v16bf join16(v8bf lo, v8bf hi) {
    union { v16bf v; v8bf h[2]; } u;
    u.h[0] = lo; u.h[1] = hi;
    return u.v;
}

// ---- Tensor Data Mover: 2D tile load Global->LDS (D# per ISA 08 §8) ----
// tile == tensor (no OOB), row-contiguous pack into LDS.
// This toolchain exposes the 6-arg builtin form: (g0, g1, g2, g3, g4, cpol).
static __device__ __forceinline__ void tdm_load_2d(const void* gptr, void* lptr,
                                                   unsigned td0, unsigned td1,
                                                   unsigned stride_elems, unsigned dsz_code) {
    unsigned long long ga = (unsigned long long)(uintptr_t)gptr;
    unsigned int lds = (unsigned int)(uintptr_t)lptr;   // LDS aperture: low 32 bits = LDS byte offset
    v4u g0;
    g0[0] = 1u;                                         // count=1, user descriptor
    g0[1] = lds;                                        // lds_addr
    g0[2] = (unsigned int)ga;                           // global_addr[31:0]
    g0[3] = (unsigned int)(ga >> 32) | (2u << 30);      // global_addr[56:32] | type=2
    v8i g1;
    g1[0] = (int)(dsz_code << 16);                      // workgroup_mask=0, data_size
    g1[1] = (int)((td0 & 0xFFFFu) << 16);               // tensor_dim0[15:0] @ bits 63:48
    g1[2] = (int)((td0 >> 16) | ((td1 & 0xFFFFu) << 16)); // tensor_dim0 hi | tensor_dim1 lo
    g1[3] = (int)((td1 >> 16) | (td0 << 16));           // tensor_dim1 hi | tile_dim0
    g1[4] = (int)(td1 & 0xFFFFu);                       // tile_dim1 (tile_dim2=0)
    g1[5] = (int)stride_elems;                          // tensor_dim0_stride[31:0]
    g1[6] = 0;
    g1[7] = 0;
    v4i z4 = {0, 0, 0, 0};
    v8i z8 = {0, 0, 0, 0, 0, 0, 0, 0};
    __builtin_amdgcn_tensor_load_to_lds(g0, g1, z4, z4, z8, 0);
}

// ---------------- f32 -> bf16 convert ----------------
__global__ void cvt_bf16_kernel(const float* __restrict__ src, __bf16* __restrict__ dst, int n) {
    int i = blockIdx.x * 256 + threadIdx.x;
    if (i < n) dst[i] = (__bf16)src[i];
}

// ---------------- film @ adaW.T + b (both adaLN blocks) ----------------
__global__ void film_gemm_kernel(const float* __restrict__ film,
                                 const float* __restrict__ W1, const float* __restrict__ b1,
                                 const float* __restrict__ W2, const float* __restrict__ b2,
                                 float* __restrict__ gb1, float* __restrict__ gb2) {
    int idx = blockIdx.x * 256 + threadIdx.x;          // 2 * 8 * 768 = 12288 exact
    int which = idx / (BB * 2 * DD);
    int rem   = idx % (BB * 2 * DD);
    int b = rem / (2 * DD);
    int o = rem % (2 * DD);
    const float* W = which ? W2 : W1;
    const float* bi = which ? b2 : b1;
    const float* f = film + b * FDIM;
    float acc = bi[o];
    const float* wr = W + (size_t)o * FDIM;
    for (int i = 0; i < FDIM; ++i) acc += f[i] * wr[i];
    (which ? gb2 : gb1)[b * 2 * DD + o] = acc;
}

// ---------------- GRN: per-(b,d) L2 over spatial ----------------
__global__ void grn_stats_kernel(const float* __restrict__ x, float* __restrict__ gx) {
    int bd = blockIdx.x;                                // BB*DD blocks
    const float* p = x + (size_t)bd * NPIX;
    float a = 0.f;
    for (int i = threadIdx.x; i < NPIX; i += 256) { float v = p[i]; a += v * v; }
    __shared__ float s[256];
    s[threadIdx.x] = a; __syncthreads();
    for (int st = 128; st > 0; st >>= 1) {
        if (threadIdx.x < st) s[threadIdx.x] += s[threadIdx.x + st];
        __syncthreads();
    }
    if (threadIdx.x == 0) gx[bd] = sqrtf(s[0]);
}

// ---------------- GRN: nx = gx / (mean_d gx + eps) ----------------
__global__ void grn_norm_kernel(const float* __restrict__ gx, float* __restrict__ nx) {
    int b = blockIdx.x;                                 // BB blocks, 384 threads
    int t = threadIdx.x;
    __shared__ float s[512];
    float v = gx[b * DD + t];
    s[t] = v;
    if (t < 128) s[DD + t] = 0.f;
    __syncthreads();
    for (int st = 256; st > 0; st >>= 1) {
        if (t < st) s[t] += s[t + st];
        __syncthreads();
    }
    float mean = s[0] / (float)DD;
    nx[b * DD + t] = v / (mean + 1e-6f);
}

// ---------------- fused adaLN(GRN) + gather + bf16 pack ----------------
template <bool PERM>
__global__ void adaln_gather_kernel(const float* __restrict__ src, const int* __restrict__ perm,
                                    const float* __restrict__ nx,
                                    const float* __restrict__ gg, const float* __restrict__ gb,
                                    const float* __restrict__ filmgb, __bf16* __restrict__ Aout) {
    int idx = blockIdx.x * 256 + threadIdx.x;           // MROW*DD exact
    int d = idx % DD;
    int m = idx / DD;
    int b = m >> 12;
    int p = m & (NPIX - 1);
    int pix = PERM ? perm[p] : p;
    float v = src[(((size_t)b * DD + d) << 12) + pix];
    float g = (1.f + gg[d]) * v * nx[b * DD + d] + gb[d] + v;           // GRN
    float y = g * (1.f + filmgb[b * 2 * DD + d]) + filmgb[b * 2 * DD + DD + d];
    Aout[idx] = (__bf16)y;
}

// ---------------- WMMA bf16 GEMM: C = A[M,K] @ W[N,K]^T, fused epilogues ----
// Wave tile 32(M)x64(N): 2 A-frags x 4 B-frags -> 8 WMMA per K-step (~21 FLOP/B).
// Block = 8 waves as 4(M) x 2(N) -> 128x128 tile.
// MODE 0: out = acc + bias               (f32)
// MODE 1: out = silu(acc + bias)         (bf16, feeds next GEMM)
// MODE 2: wi  = image + (acc+bias)*s1    scattered via perm (cloud un-shuffle)
// MODE 3: out = image + (wi + (acc+bias)*s1) * s2
template <int MODE>
__global__ __launch_bounds__(256)
void wmma_gemm_kernel(const __bf16* __restrict__ A, const __bf16* __restrict__ W,
                      const float* __restrict__ bias,
                      float* __restrict__ outF, __bf16* __restrict__ outB,
                      const float* __restrict__ image, const float* __restrict__ wi,
                      const int* __restrict__ perm,
                      const float* __restrict__ s1, const float* __restrict__ s2,
                      int M, int N, int K) {
    const int lane  = threadIdx.x & 31;
    const int wave  = threadIdx.x >> 5;
    const int r     = lane & 15;
    const int h     = lane >> 4;
    const int waveM = wave & 3;
    const int waveN = wave >> 2;
    const int m0 = blockIdx.y * 128 + waveM * 32;
    const int n0 = blockIdx.x * 128 + waveN * 64;

    const __bf16* A0 = A + (size_t)(m0 + r) * K;
    const __bf16* A1 = A + (size_t)(m0 + 16 + r) * K;
    const __bf16* Wr0 = W + (size_t)(n0 + r) * K;
    const __bf16* Wr1 = W + (size_t)(n0 + 16 + r) * K;
    const __bf16* Wr2 = W + (size_t)(n0 + 32 + r) * K;
    const __bf16* Wr3 = W + (size_t)(n0 + 48 + r) * K;

    v8f acc[2][4] = {};
    for (int kb = 0; kb < K; kb += 32) {
        // A frag: lane holds K in [8h,8h+8) and [16+8h,16+8h+8) of its row
        v16bf a0 = join16(*(const v8bf*)(A0 + kb + 8 * h), *(const v8bf*)(A0 + kb + 16 + 8 * h));
        v16bf a1 = join16(*(const v8bf*)(A1 + kb + 8 * h), *(const v8bf*)(A1 + kb + 16 + 8 * h));
        // B frag: lane holds K in [8h,8h+8) and [16+8h,16+8h+8) of its column row
        v16bf b0 = join16(*(const v8bf*)(Wr0 + kb + 8 * h), *(const v8bf*)(Wr0 + kb + 16 + 8 * h));
        v16bf b1 = join16(*(const v8bf*)(Wr1 + kb + 8 * h), *(const v8bf*)(Wr1 + kb + 16 + 8 * h));
        v16bf b2 = join16(*(const v8bf*)(Wr2 + kb + 8 * h), *(const v8bf*)(Wr2 + kb + 16 + 8 * h));
        v16bf b3 = join16(*(const v8bf*)(Wr3 + kb + 8 * h), *(const v8bf*)(Wr3 + kb + 16 + 8 * h));
        acc[0][0] = __builtin_amdgcn_wmma_f32_16x16x32_bf16(false, a0, false, b0, (short)0, acc[0][0], false, false);
        acc[1][0] = __builtin_amdgcn_wmma_f32_16x16x32_bf16(false, a1, false, b0, (short)0, acc[1][0], false, false);
        acc[0][1] = __builtin_amdgcn_wmma_f32_16x16x32_bf16(false, a0, false, b1, (short)0, acc[0][1], false, false);
        acc[1][1] = __builtin_amdgcn_wmma_f32_16x16x32_bf16(false, a1, false, b1, (short)0, acc[1][1], false, false);
        acc[0][2] = __builtin_amdgcn_wmma_f32_16x16x32_bf16(false, a0, false, b2, (short)0, acc[0][2], false, false);
        acc[1][2] = __builtin_amdgcn_wmma_f32_16x16x32_bf16(false, a1, false, b2, (short)0, acc[1][2], false, false);
        acc[0][3] = __builtin_amdgcn_wmma_f32_16x16x32_bf16(false, a0, false, b3, (short)0, acc[0][3], false, false);
        acc[1][3] = __builtin_amdgcn_wmma_f32_16x16x32_bf16(false, a1, false, b3, (short)0, acc[1][3], false, false);
    }

#pragma unroll
    for (int ni = 0; ni < 4; ++ni) {
        const int n = n0 + ni * 16 + r;
        const float bi = bias[n];
        const float cs = (MODE == 2 || MODE == 3) ? s1[n] : 0.f;
        const float fs = (MODE == 3) ? s2[n] : 0.f;
#pragma unroll
        for (int mi = 0; mi < 2; ++mi) {
#pragma unroll
            for (int j = 0; j < 8; ++j) {
                const int m = m0 + mi * 16 + 8 * h + j;
                float v = acc[mi][ni][j] + bi;
                if (MODE == 0) {
                    outF[(size_t)m * N + n] = v;
                } else if (MODE == 1) {
                    outB[(size_t)m * N + n] = (__bf16)(v / (1.f + __expf(-v)));
                } else if (MODE == 2) {
                    int b = m >> 12, p = m & (NPIX - 1);
                    int pix = perm[p];
                    size_t i0 = (((size_t)b * DD + n) << 12) + pix;
                    outF[i0] = image[i0] + v * cs;
                } else { // MODE 3
                    int b = m >> 12, p = m & (NPIX - 1);
                    size_t i0 = (((size_t)b * DD + n) << 12) + p;
                    outF[i0] = image[i0] + (wi[i0] + v * cs) * fs;
                }
            }
        }
    }
}

// ---------------- per-row LayerNorm + elu + 1 (in place) ----------------
__global__ void ln_elu_kernel(float* __restrict__ x, const float* __restrict__ g,
                              const float* __restrict__ b) {
    int row = blockIdx.x;                               // MROW blocks, 384 threads
    int t = threadIdx.x;
    float* xr = x + (size_t)row * DD;
    float v = xr[t];
    __shared__ float s1[512], s2[512];
    s1[t] = v; s2[t] = v * v;
    if (t < 128) { s1[DD + t] = 0.f; s2[DD + t] = 0.f; }
    __syncthreads();
    for (int st = 256; st > 0; st >>= 1) {
        if (t < st) { s1[t] += s1[t + st]; s2[t] += s2[t + st]; }
        __syncthreads();
    }
    float mean = s1[0] / (float)DD;
    float var  = s2[0] / (float)DD - mean * mean;
    float y = (v - mean) * rsqrtf(var + 1e-5f) * g[t] + b[t];
    y = (y > 0.f) ? y : (__expf(y) - 1.f);              // elu
    xr[t] = y + 1.f;
}

// ---- kv = k^T v and ksum per (cloud,head): TDM-staged LDS tiles ----------
__global__ __launch_bounds__(256)
void kv_kernel(const float* __restrict__ K, const float* __restrict__ V,
               float* __restrict__ kv, float* __restrict__ ksum) {
    int bh = blockIdx.x;                                // 32*8 blocks
    int bc = bh >> 3, hh = bh & 7;
    int t = threadIdx.x;
    __shared__ float sk[128 * HDD];
    __shared__ float sv[128 * HDD];
    float acc[9];
#pragma unroll
    for (int u = 0; u < 9; ++u) acc[u] = 0.f;
    int du[9], eu[9];
#pragma unroll
    for (int u = 0; u < 9; ++u) { int o = t + u * 256; du[u] = o / HDD; eu[u] = o % HDD; }
    float ksacc = 0.f;
    for (int nb = 0; nb < GSZ; nb += 128) {
        // TDM: DMA [128 x 48] f32 tiles (row stride DD) into LDS; wave 0 issues.
        size_t srcbase = ((size_t)(bc * GSZ + nb)) * DD + hh * HDD;
        if (t < 32) {
            tdm_load_2d(K + srcbase, sk, HDD, 128, DD, 2u /*4-byte*/);
            tdm_load_2d(V + srcbase, sv, HDD, 128, DD, 2u);
            __builtin_amdgcn_s_wait_tensorcnt(0);
        }
        __syncthreads();
        for (int n = 0; n < 128; ++n) {
#pragma unroll
            for (int u = 0; u < 9; ++u)
                acc[u] += sk[n * HDD + du[u]] * sv[n * HDD + eu[u]];
            if (t < HDD) ksacc += sk[n * HDD + t];
        }
        __syncthreads();
    }
#pragma unroll
    for (int u = 0; u < 9; ++u) kv[(size_t)bh * HDD * HDD + t + u * 256] = acc[u];
    if (t < HDD) ksum[bh * HDD + t] = ksacc;
}

// ---------------- attention combine: (q@kv)/(q.ksum) -> bf16 A for Wo ------
__global__ void attn_out_kernel(const float* __restrict__ Q, const float* __restrict__ kv,
                                const float* __restrict__ ksum, __bf16* __restrict__ AO) {
    int row = blockIdx.x;                               // MROW blocks, 384 threads
    int bc = row >> 10;
    int t = threadIdx.x;
    __shared__ float sq[DD];
    sq[t] = Q[(size_t)row * DD + t];
    __syncthreads();
    int hh = t / HDD, e = t % HDD;
    const float* kvp = kv + (size_t)(bc * NHD + hh) * HDD * HDD;
    const float* ksp = ksum + (size_t)(bc * NHD + hh) * HDD;
    float num = 0.f, den = 0.f;
#pragma unroll 4
    for (int dd = 0; dd < HDD; ++dd) {
        float qv = sq[hh * HDD + dd];
        num += qv * kvp[dd * HDD + e];
        den += qv * ksp[dd];
    }
    AO[(size_t)row * DD + t] = (__bf16)(num / (den + 1e-8f));
}

// ---------------- host side ----------------
extern "C" void kernel_launch(void* const* d_in, const int* in_sizes, int n_in,
                              void* d_out, int out_size, void* d_ws, size_t ws_size,
                              hipStream_t stream) {
    const float* image  = (const float*)d_in[0];
    const float* film   = (const float*)d_in[1];
    const int*   perm   = (const int*)  d_in[2];
    const float* Wq = (const float*)d_in[3];  const float* bq = (const float*)d_in[4];
    const float* Wk = (const float*)d_in[5];  const float* bk = (const float*)d_in[6];
    const float* Wv = (const float*)d_in[7];  const float* bv = (const float*)d_in[8];
    const float* Wo = (const float*)d_in[9];  const float* bo = (const float*)d_in[10];
    const float* lnq_g = (const float*)d_in[11]; const float* lnq_b = (const float*)d_in[12];
    const float* lnk_g = (const float*)d_in[13]; const float* lnk_b = (const float*)d_in[14];
    const float* ada1W = (const float*)d_in[15]; const float* ada1b = (const float*)d_in[16];
    const float* grn1g = (const float*)d_in[17]; const float* grn1b = (const float*)d_in[18];
    const float* ada2W = (const float*)d_in[19]; const float* ada2b = (const float*)d_in[20];
    const float* grn2g = (const float*)d_in[21]; const float* grn2b = (const float*)d_in[22];
    const float* c1W = (const float*)d_in[23]; const float* c1b = (const float*)d_in[24];
    const float* c2W = (const float*)d_in[25]; const float* c2b = (const float*)d_in[26];
    const float* clouds = (const float*)d_in[27];
    const float* ffns   = (const float*)d_in[28];
    const float* finals = (const float*)d_in[29];

    char* ws = (char*)d_ws;
    size_t off = 0;
    auto take = [&](size_t bytes) { size_t o = off; off += (bytes + 255) & ~(size_t)255; return o; };
    float*  gb1  = (float*)(ws + take(BB * 2 * DD * 4));
    float*  gb2  = (float*)(ws + take(BB * 2 * DD * 4));
    float*  gx   = (float*)(ws + take(BB * DD * 4));
    float*  nx   = (float*)(ws + take(BB * DD * 4));
    __bf16* wqb  = (__bf16*)(ws + take((size_t)DD * DD * 2));
    __bf16* wkb  = (__bf16*)(ws + take((size_t)DD * DD * 2));
    __bf16* wvb  = (__bf16*)(ws + take((size_t)DD * DD * 2));
    __bf16* wob  = (__bf16*)(ws + take((size_t)DD * DD * 2));
    __bf16* w1b  = (__bf16*)(ws + take((size_t)DFF * DD * 2));
    __bf16* w2b  = (__bf16*)(ws + take((size_t)DD * DFF * 2));
    __bf16* Abf  = (__bf16*)(ws + take((size_t)MROW * DD * 2));
    float*  Qf   = (float*)(ws + take((size_t)MROW * DD * 4));
    float*  Kf   = (float*)(ws + take((size_t)MROW * DD * 4));
    float*  Vf   = (float*)(ws + take((size_t)MROW * DD * 4));
    float*  kvb  = (float*)(ws + take((size_t)32 * NHD * HDD * HDD * 4));
    float*  ksb  = (float*)(ws + take((size_t)32 * NHD * HDD * 4));
    __bf16* AObf = (__bf16*)(ws + take((size_t)MROW * DD * 2));
    float*  WIf  = (float*)(ws + take((size_t)MROW * DD * 4));
    __bf16* H1bf = (__bf16*)Qf;   // reuse Q+K region (exactly MROW*DFF*2 bytes)
    float*  out  = (float*)d_out;

    // 1) weights -> bf16
    cvt_bf16_kernel<<<(DD * DD + 255) / 256, 256, 0, stream>>>(Wq, wqb, DD * DD);
    cvt_bf16_kernel<<<(DD * DD + 255) / 256, 256, 0, stream>>>(Wk, wkb, DD * DD);
    cvt_bf16_kernel<<<(DD * DD + 255) / 256, 256, 0, stream>>>(Wv, wvb, DD * DD);
    cvt_bf16_kernel<<<(DD * DD + 255) / 256, 256, 0, stream>>>(Wo, wob, DD * DD);
    cvt_bf16_kernel<<<(DFF * DD + 255) / 256, 256, 0, stream>>>(c1W, w1b, DFF * DD);
    cvt_bf16_kernel<<<(DD * DFF + 255) / 256, 256, 0, stream>>>(c2W, w2b, DD * DFF);

    // 2) film -> gamma/beta for both adaLN blocks
    film_gemm_kernel<<<(2 * BB * 2 * DD) / 256, 256, 0, stream>>>(film, ada1W, ada1b, ada2W, ada2b, gb1, gb2);

    // 3) adaLN-1 (GRN on image) + permuted gather -> bf16 A
    grn_stats_kernel<<<BB * DD, 256, 0, stream>>>(image, gx);
    grn_norm_kernel<<<BB, DD, 0, stream>>>(gx, nx);
    adaln_gather_kernel<true><<<(MROW * DD) / 256, 256, 0, stream>>>(image, perm, nx, grn1g, grn1b, gb1, Abf);

    // 4) Q/K/V projections (WMMA)
    dim3 blk(256);
    dim3 g384(DD / 128, MROW / 128);
    wmma_gemm_kernel<0><<<g384, blk, 0, stream>>>(Abf, wqb, bq, Qf, nullptr, nullptr, nullptr, nullptr, nullptr, nullptr, MROW, DD, DD);
    wmma_gemm_kernel<0><<<g384, blk, 0, stream>>>(Abf, wkb, bk, Kf, nullptr, nullptr, nullptr, nullptr, nullptr, nullptr, MROW, DD, DD);
    wmma_gemm_kernel<0><<<g384, blk, 0, stream>>>(Abf, wvb, bv, Vf, nullptr, nullptr, nullptr, nullptr, nullptr, nullptr, MROW, DD, DD);

    // 5) LN + elu + 1 on q,k
    ln_elu_kernel<<<MROW, DD, 0, stream>>>(Qf, lnq_g, lnq_b);
    ln_elu_kernel<<<MROW, DD, 0, stream>>>(Kf, lnk_g, lnk_b);

    // 6) linear attention core (TDM-staged LDS tiles)
    kv_kernel<<<32 * NHD, 256, 0, stream>>>(Kf, Vf, kvb, ksb);
    attn_out_kernel<<<MROW, DD, 0, stream>>>(Qf, kvb, ksb, AObf);

    // 7) Wo projection + un-permute scatter + residual scale -> wi
    wmma_gemm_kernel<2><<<g384, blk, 0, stream>>>(AObf, wob, bo, WIf, nullptr, image, nullptr, perm, clouds, nullptr, MROW, DD, DD);

    // 8) adaLN-2 (GRN on wi), no permutation -> bf16 A
    grn_stats_kernel<<<BB * DD, 256, 0, stream>>>(WIf, gx);
    grn_norm_kernel<<<BB, DD, 0, stream>>>(gx, nx);
    adaln_gather_kernel<false><<<(MROW * DD) / 256, 256, 0, stream>>>(WIf, nullptr, nx, grn2g, grn2b, gb2, Abf);

    // 9) FFN: conv1 (+SiLU -> bf16), conv2 (+final combine -> d_out)
    dim3 g1536(DFF / 128, MROW / 128);
    wmma_gemm_kernel<1><<<g1536, blk, 0, stream>>>(Abf, w1b, c1b, nullptr, H1bf, nullptr, nullptr, nullptr, nullptr, nullptr, MROW, DFF, DD);
    wmma_gemm_kernel<3><<<g384, blk, 0, stream>>>(H1bf, w2b, c2b, out, nullptr, image, WIf, nullptr, ffns, finals, MROW, DD, DFF);
}
